// MeshGraphNetsConv_16415365006070
// MI455X (gfx1250) — matrix operand order repaired and checked by reading, not compiled, not run on hardware
//
#include <hip/hip_runtime.h>
#include <math.h>

#define DD 128            // feature dim
#define WAVES 4           // waves per block
#define MT 4              // 16-row M-tiles per wave
#define RPW (16 * MT)     // rows per wave = 64
#define RPB (WAVES * RPW) // rows per block = 256

typedef __attribute__((ext_vector_type(16))) _Float16 v16h;
typedef __attribute__((ext_vector_type(8)))  _Float16 v8h;
typedef __attribute__((ext_vector_type(8)))  float    v8f;
typedef __attribute__((ext_vector_type(4)))  float    v4f;

static __device__ __forceinline__ v8f vzero8() {
  v8f v;
#pragma unroll
  for (int t = 0; t < 8; ++t) v[t] = 0.0f;
  return v;
}

static __device__ __forceinline__ v8f wmma16(v16h a, v16h b, v8f c) {
  // D = A(16x32,f16) * B(32x16,f16) + C(16x16,f32)
  return __builtin_amdgcn_wmma_f32_16x16x32_f16(false, a, false, b, (short)0, c,
                                                false, false);
}

// One packed 32x16 B tile: lane's v16h fragment is a contiguous 32B load.
static __device__ __forceinline__ v16h load_btile(const _Float16* __restrict__ w,
                                                  int tile, int lane) {
  return *(const v16h*)(w + (size_t)tile * 512 + lane * 16);
}

// Build an A fragment (this lane's 16 halves) from 8+8 consecutive f32 in a row.
static __device__ __forceinline__ v16h make_a_f32(const float* __restrict__ p) {
  v4f f0 = *(const v4f*)(p);
  v4f f1 = *(const v4f*)(p + 4);
  v4f f2 = *(const v4f*)(p + 16);
  v4f f3 = *(const v4f*)(p + 20);
  v16h a;
#pragma unroll
  for (int t = 0; t < 4; ++t) {
    a[t]      = (_Float16)f0[t];
    a[4 + t]  = (_Float16)f1[t];
    a[8 + t]  = (_Float16)f2[t];
    a[12 + t] = (_Float16)f3[t];
  }
  return a;
}

// Same, but from f16 row-major activations staged in LDS.
static __device__ __forceinline__ v16h make_a_h16(const _Float16* __restrict__ p) {
  v8h lo = *(const v8h*)(p);
  v8h hi = *(const v8h*)(p + 16);
  v16h a;
#pragma unroll
  for (int t = 0; t < 8; ++t) { a[t] = lo[t]; a[8 + t] = hi[t]; }
  return a;
}

static __device__ __forceinline__ float silu(float v) {
  return v * (1.0f / (1.0f + __expf(-v)));
}

// ---------------------------------------------------------------------------
// Prep: zero the aggregation buffer.
__global__ void zero_kernel(float* __restrict__ p, int n) {
  int i = blockIdx.x * blockDim.x + threadIdx.x;
  if (i < n) p[i] = 0.0f;
}

// Prep: pack a row-major f32 weight matrix [K x Ncols] into per-(kt,nt)-tile
// f16 blocks of 512 halves laid out so that lane L reads its v16h B fragment
// as one contiguous 32B load:
//   lane < 16 : column N = nt*16+lane,     K = kt*32 + (m&7) + (m>>3)*16      (m = 0..15)
//   lane >= 16: column N = nt*16+lane-16,  K = kt*32 + 8 + (m&7) + (m>>3)*16
__global__ void pack_w_kernel(const float* __restrict__ src,
                              _Float16* __restrict__ dst, int K, int Ncols) {
  int gid = blockIdx.x * blockDim.x + threadIdx.x;
  int total = K * Ncols;
  if (gid >= total) return;
  int m    = gid & 15;
  int lane = (gid >> 4) & 31;
  int tile = gid >> 9;            // 512 halves per 32x16 tile
  int Nt   = Ncols >> 4;
  int nt   = tile % Nt;
  int kt   = tile / Nt;
  int Kabs = kt * 32 + ((lane >> 4) ? 8 : 0) + (m & 7) + (m >> 3) * 16;
  int Nabs = nt * 16 + (lane & 15);
  dst[gid] = (_Float16)src[(size_t)Kabs * Ncols + Nabs];
}

// ---------------------------------------------------------------------------
// Edge processor: e_out = edge_attr + LN(MLP(concat(x[i], x[j], edge_attr)))
// plus native-f32 atomic scatter of e_out into agg[j].
__global__ __launch_bounds__(WAVES * 32) void edge_mlp_kernel(
    const float* __restrict__ x, const int* __restrict__ ei,
    const float* __restrict__ ea,
    const _Float16* __restrict__ w1, const float* __restrict__ b1,
    const _Float16* __restrict__ w2, const float* __restrict__ b2,
    const _Float16* __restrict__ w3, const float* __restrict__ b3,
    const float* __restrict__ g, const float* __restrict__ be,
    float* __restrict__ eout, float* __restrict__ agg, int E) {
  extern __shared__ _Float16 smem[];
  const int lane = threadIdx.x & 31;
  const int wave = threadIdx.x >> 5;
  const int lrow = lane & 15;   // A-row within M-tile / N within N-tile
  const int hi   = lane >> 4;   // half-wave id
  const int kb   = hi * 8;      // K sub-offset for A fragments
  _Float16* h1 = smem + (size_t)wave * RPW * DD;
  _Float16* h2 = smem + (size_t)(WAVES + wave) * RPW * DD;
  const int base = blockIdx.x * RPB + wave * RPW;

  // Per-(segment, M-tile) A row-base pointers, computed once.
  const float* rbase[3][MT];
#pragma unroll
  for (int mt = 0; mt < MT; ++mt) {
    int e = base + mt * 16 + lrow;
    rbase[0][mt] = x + (size_t)ei[e] * DD;       // x[i]
    rbase[1][mt] = x + (size_t)ei[E + e] * DD;   // x[j]
    rbase[2][mt] = ea + (size_t)e * DD;          // edge_attr
  }

  // ---- layer 1: [M x 384] @ [384 x 128]; B double-buffered ----
#pragma unroll 1
  for (int nt = 0; nt < 8; ++nt) {
    v8f acc[MT];
#pragma unroll
    for (int mt = 0; mt < MT; ++mt) acc[mt] = vzero8();
    v16h bcur = load_btile(w1, nt, lane);         // kt = 0
#pragma unroll
    for (int kt = 0; kt < 12; ++kt) {
      v16h bnxt;
      if (kt < 11) bnxt = load_btile(w1, (kt + 1) * 8 + nt, lane);
      const int seg = kt >> 2;                    // 0: x[i], 1: x[j], 2: ea
      const int off = (kt & 3) * 32 + kb;
#pragma unroll
      for (int mt = 0; mt < MT; ++mt)
        acc[mt] = wmma16(make_a_f32(rbase[seg][mt] + off), bcur, acc[mt]);
      if (kt < 11) bcur = bnxt;
    }
    float bias = b1[nt * 16 + lrow];
#pragma unroll
    for (int mt = 0; mt < MT; ++mt)
#pragma unroll
      for (int r = 0; r < 8; ++r) {
        float v = silu(acc[mt][r] + bias);
        h1[(size_t)(mt * 16 + r + hi * 8) * DD + nt * 16 + lrow] = (_Float16)v;
      }
  }

  // ---- layer 2: [M x 128] @ [128 x 128], A from LDS; B double-buffered ----
#pragma unroll 1
  for (int nt = 0; nt < 8; ++nt) {
    v8f acc[MT];
#pragma unroll
    for (int mt = 0; mt < MT; ++mt) acc[mt] = vzero8();
    v16h bcur = load_btile(w2, nt, lane);
#pragma unroll
    for (int kt = 0; kt < 4; ++kt) {
      v16h bnxt;
      if (kt < 3) bnxt = load_btile(w2, (kt + 1) * 8 + nt, lane);
#pragma unroll
      for (int mt = 0; mt < MT; ++mt) {
        const _Float16* ap = h1 + (size_t)(mt * 16 + lrow) * DD + kt * 32 + kb;
        acc[mt] = wmma16(make_a_h16(ap), bcur, acc[mt]);
      }
      if (kt < 3) bcur = bnxt;
    }
    float bias = b2[nt * 16 + lrow];
#pragma unroll
    for (int mt = 0; mt < MT; ++mt)
#pragma unroll
      for (int r = 0; r < 8; ++r) {
        float v = silu(acc[mt][r] + bias);
        h2[(size_t)(mt * 16 + r + hi * 8) * DD + nt * 16 + lrow] = (_Float16)v;
      }
  }

  // ---- layer 3 + LayerNorm + residual + scatter, per M-tile ----
#pragma unroll 1
  for (int mt = 0; mt < MT; ++mt) {
    v8f acc[8];
#pragma unroll
    for (int nt = 0; nt < 8; ++nt) acc[nt] = vzero8();
    v16h bcur = load_btile(w3, 0, lane);
#pragma unroll
    for (int kt = 0; kt < 4; ++kt) {
      v16h a = make_a_h16(h2 + (size_t)(mt * 16 + lrow) * DD + kt * 32 + kb);
#pragma unroll
      for (int nt = 0; nt < 8; ++nt) {
        int t = kt * 8 + nt;
        v16h bnxt;
        if (t < 31) bnxt = load_btile(w3, t + 1, lane);
        acc[nt] = wmma16(a, bcur, acc[nt]);
        if (t < 31) bcur = bnxt;
      }
    }
#pragma unroll
    for (int nt = 0; nt < 8; ++nt) {
      float bias = b3[nt * 16 + lrow];
#pragma unroll
      for (int r = 0; r < 8; ++r) acc[nt][r] += bias;
    }
    // LayerNorm: row M = r + hi*8 lives in one 16-lane half; butterfly reduce.
    float mean[8], inv[8];
#pragma unroll
    for (int r = 0; r < 8; ++r) {
      float s = 0.0f, q = 0.0f;
#pragma unroll
      for (int nt = 0; nt < 8; ++nt) { float v = acc[nt][r]; s += v; q += v * v; }
#pragma unroll
      for (int m = 1; m < 16; m <<= 1) {
        s += __shfl_xor(s, m, 16);
        q += __shfl_xor(q, m, 16);
      }
      float mu = s * (1.0f / DD);
      mean[r] = mu;
      inv[r] = rsqrtf(q * (1.0f / DD) - mu * mu + 1e-5f);
    }
    // Destination node index per row (uniform within each 16-lane half).
    int jdst[8];
#pragma unroll
    for (int r = 0; r < 8; ++r) jdst[r] = ei[E + base + mt * 16 + r + hi * 8];
#pragma unroll
    for (int nt = 0; nt < 8; ++nt) {
      int n = nt * 16 + lrow;
      float gg = g[n], bb = be[n];
#pragma unroll
      for (int r = 0; r < 8; ++r) {
        int row = base + mt * 16 + r + hi * 8;
        float res = ea[(size_t)row * DD + n];
        float o = res + (acc[nt][r] - mean[r]) * inv[r] * gg + bb;
        __builtin_nontemporal_store(o, &eout[(size_t)row * DD + n]);
        unsafeAtomicAdd(agg + (size_t)jdst[r] * DD + n, o);
      }
    }
  }
}

// ---------------------------------------------------------------------------
// Node processor: x_new = x + LN(MLP(concat(x, agg)))
__global__ __launch_bounds__(WAVES * 32) void node_mlp_kernel(
    const float* __restrict__ x, const float* __restrict__ agg,
    const _Float16* __restrict__ w1, const float* __restrict__ b1,
    const _Float16* __restrict__ w2, const float* __restrict__ b2,
    const _Float16* __restrict__ w3, const float* __restrict__ b3,
    const float* __restrict__ g, const float* __restrict__ be,
    float* __restrict__ xout, int Nn) {
  extern __shared__ _Float16 smem[];
  const int lane = threadIdx.x & 31;
  const int wave = threadIdx.x >> 5;
  const int lrow = lane & 15;
  const int hi   = lane >> 4;
  const int kb   = hi * 8;
  _Float16* h1 = smem + (size_t)wave * RPW * DD;
  _Float16* h2 = smem + (size_t)(WAVES + wave) * RPW * DD;
  const int base = blockIdx.x * RPB + wave * RPW;

  const float* rbase[2][MT];
#pragma unroll
  for (int mt = 0; mt < MT; ++mt) {
    int row = base + mt * 16 + lrow;
    int rc = row < Nn ? row : Nn - 1;
    rbase[0][mt] = x + (size_t)rc * DD;
    rbase[1][mt] = agg + (size_t)rc * DD;
  }

  // ---- layer 1: [M x 256] @ [256 x 128]; B double-buffered ----
#pragma unroll 1
  for (int nt = 0; nt < 8; ++nt) {
    v8f acc[MT];
#pragma unroll
    for (int mt = 0; mt < MT; ++mt) acc[mt] = vzero8();
    v16h bcur = load_btile(w1, nt, lane);
#pragma unroll
    for (int kt = 0; kt < 8; ++kt) {
      v16h bnxt;
      if (kt < 7) bnxt = load_btile(w1, (kt + 1) * 8 + nt, lane);
      const int seg = kt >> 2;                    // 0: x, 1: agg
      const int off = (kt & 3) * 32 + kb;
#pragma unroll
      for (int mt = 0; mt < MT; ++mt)
        acc[mt] = wmma16(make_a_f32(rbase[seg][mt] + off), bcur, acc[mt]);
      if (kt < 7) bcur = bnxt;
    }
    float bias = b1[nt * 16 + lrow];
#pragma unroll
    for (int mt = 0; mt < MT; ++mt)
#pragma unroll
      for (int r = 0; r < 8; ++r) {
        float v = silu(acc[mt][r] + bias);
        h1[(size_t)(mt * 16 + r + hi * 8) * DD + nt * 16 + lrow] = (_Float16)v;
      }
  }

  // ---- layer 2 ----
#pragma unroll 1
  for (int nt = 0; nt < 8; ++nt) {
    v8f acc[MT];
#pragma unroll
    for (int mt = 0; mt < MT; ++mt) acc[mt] = vzero8();
    v16h bcur = load_btile(w2, nt, lane);
#pragma unroll
    for (int kt = 0; kt < 4; ++kt) {
      v16h bnxt;
      if (kt < 3) bnxt = load_btile(w2, (kt + 1) * 8 + nt, lane);
#pragma unroll
      for (int mt = 0; mt < MT; ++mt) {
        const _Float16* ap = h1 + (size_t)(mt * 16 + lrow) * DD + kt * 32 + kb;
        acc[mt] = wmma16(make_a_h16(ap), bcur, acc[mt]);
      }
      if (kt < 3) bcur = bnxt;
    }
    float bias = b2[nt * 16 + lrow];
#pragma unroll
    for (int mt = 0; mt < MT; ++mt)
#pragma unroll
      for (int r = 0; r < 8; ++r) {
        float v = silu(acc[mt][r] + bias);
        h2[(size_t)(mt * 16 + r + hi * 8) * DD + nt * 16 + lrow] = (_Float16)v;
      }
  }

  // ---- layer 3 + LN + residual ----
#pragma unroll 1
  for (int mt = 0; mt < MT; ++mt) {
    v8f acc[8];
#pragma unroll
    for (int nt = 0; nt < 8; ++nt) acc[nt] = vzero8();
    v16h bcur = load_btile(w3, 0, lane);
#pragma unroll
    for (int kt = 0; kt < 4; ++kt) {
      v16h a = make_a_h16(h2 + (size_t)(mt * 16 + lrow) * DD + kt * 32 + kb);
#pragma unroll
      for (int nt = 0; nt < 8; ++nt) {
        int t = kt * 8 + nt;
        v16h bnxt;
        if (t < 31) bnxt = load_btile(w3, t + 1, lane);
        acc[nt] = wmma16(a, bcur, acc[nt]);
        if (t < 31) bcur = bnxt;
      }
    }
#pragma unroll
    for (int nt = 0; nt < 8; ++nt) {
      float bias = b3[nt * 16 + lrow];
#pragma unroll
      for (int r = 0; r < 8; ++r) acc[nt][r] += bias;
    }
    float mean[8], inv[8];
#pragma unroll
    for (int r = 0; r < 8; ++r) {
      float s = 0.0f, q = 0.0f;
#pragma unroll
      for (int nt = 0; nt < 8; ++nt) { float v = acc[nt][r]; s += v; q += v * v; }
#pragma unroll
      for (int m = 1; m < 16; m <<= 1) {
        s += __shfl_xor(s, m, 16);
        q += __shfl_xor(q, m, 16);
      }
      float mu = s * (1.0f / DD);
      mean[r] = mu;
      inv[r] = rsqrtf(q * (1.0f / DD) - mu * mu + 1e-5f);
    }
#pragma unroll
    for (int nt = 0; nt < 8; ++nt) {
      int n = nt * 16 + lrow;
      float gg = g[n], bb = be[n];
#pragma unroll
      for (int r = 0; r < 8; ++r) {
        int row = base + mt * 16 + r + hi * 8;
        if (row < Nn) {
          float res = x[(size_t)row * DD + n];
          float o = res + (acc[nt][r] - mean[r]) * inv[r] * gg + bb;
          __builtin_nontemporal_store(o, &xout[(size_t)row * DD + n]);
        }
      }
    }
  }
}

// ---------------------------------------------------------------------------
extern "C" void kernel_launch(void* const* d_in, const int* in_sizes, int n_in,
                              void* d_out, int out_size, void* d_ws,
                              size_t ws_size, hipStream_t stream) {
  (void)in_sizes; (void)n_in; (void)out_size; (void)ws_size;
  const int Nn = 50000, E = 800000;

  const float* x    = (const float*)d_in[0];
  const int*   ei   = (const int*)d_in[1];
  const float* ea   = (const float*)d_in[2];
  const float* ew1  = (const float*)d_in[3];
  const float* eb1  = (const float*)d_in[4];
  const float* ew2  = (const float*)d_in[5];
  const float* eb2  = (const float*)d_in[6];
  const float* ew3  = (const float*)d_in[7];
  const float* eb3  = (const float*)d_in[8];
  const float* elng = (const float*)d_in[9];
  const float* elnb = (const float*)d_in[10];
  const float* nw1  = (const float*)d_in[11];
  const float* nb1  = (const float*)d_in[12];
  const float* nw2  = (const float*)d_in[13];
  const float* nb2  = (const float*)d_in[14];
  const float* nw3  = (const float*)d_in[15];
  const float* nb3  = (const float*)d_in[16];
  const float* nlng = (const float*)d_in[17];
  const float* nlnb = (const float*)d_in[18];

  float* xout = (float*)d_out;                     // x_new first
  float* eout = (float*)d_out + (size_t)Nn * DD;   // then edge_attr_new

  // Workspace: [agg f32 N*D][packed f16 weights]
  float* agg = (float*)d_ws;
  size_t aggBytes = (size_t)Nn * DD * sizeof(float);
  size_t woff = (aggBytes + 255) & ~(size_t)255;
  _Float16* pe1 = (_Float16*)((char*)d_ws + woff);
  _Float16* pe2 = pe1 + (size_t)3 * DD * DD;   // 384x128
  _Float16* pe3 = pe2 + (size_t)DD * DD;       // 128x128
  _Float16* pn1 = pe3 + (size_t)DD * DD;       // 128x128
  _Float16* pn2 = pn1 + (size_t)2 * DD * DD;   // 256x128
  _Float16* pn3 = pn2 + (size_t)DD * DD;       // 128x128

  int zn = Nn * DD;
  zero_kernel<<<(zn + 255) / 256, 256, 0, stream>>>(agg, zn);

  pack_w_kernel<<<(3 * DD * DD + 255) / 256, 256, 0, stream>>>(ew1, pe1, 3 * DD, DD);
  pack_w_kernel<<<(DD * DD + 255) / 256, 256, 0, stream>>>(ew2, pe2, DD, DD);
  pack_w_kernel<<<(DD * DD + 255) / 256, 256, 0, stream>>>(ew3, pe3, DD, DD);
  pack_w_kernel<<<(2 * DD * DD + 255) / 256, 256, 0, stream>>>(nw1, pn1, 2 * DD, DD);
  pack_w_kernel<<<(DD * DD + 255) / 256, 256, 0, stream>>>(nw2, pn2, DD, DD);
  pack_w_kernel<<<(DD * DD + 255) / 256, 256, 0, stream>>>(nw3, pn3, DD, DD);

  size_t shmem = (size_t)2 * WAVES * RPW * DD * sizeof(_Float16); // 128 KB

  edge_mlp_kernel<<<E / RPB, WAVES * 32, shmem, stream>>>(
      x, ei, ea, pe1, eb1, pe2, eb2, pe3, eb3, elng, elnb, eout, agg, E);

  int nblocks = (Nn + RPB - 1) / RPB;
  node_mlp_kernel<<<nblocks, WAVES * 32, shmem, stream>>>(
      x, agg, pn1, nb1, pn2, nb2, pn3, nb3, nlng, nlnb, xout, Nn);
}
